// RecurrentGCN_62775241998691
// MI455X (gfx1250) — compile-verified, matrix operand order checked by reading, not run on hardware
//
#include <hip/hip_runtime.h>
#include <hip/hip_bf16.h>

typedef __attribute__((ext_vector_type(16))) _Float16 v16h;
typedef __attribute__((ext_vector_type(8)))  float    v8f;

#define N_NODES   512000
#define N_TILES   32000     // N_NODES / 16
#define TPW       5         // tiles per wave (125 tiles/graph = 25 waves/graph)
#define N_WAVES   6400      // N_TILES / TPW
#define WAVES_PG  25        // waves per graph
#define N_GRAPH   256
#define F_IN      64
#define HID       32

// d_ws layout (bytes):
//   [0,    8192)  B fragments, f16, WMMA register order:
//                 frag[gate][t][kt][lane][i], gate0=z gate1=h,
//                 t = N-tile (0..1), kt = K-tile (0..1), lane 0..31, i 0..15
//   [4096, 4224)  bz f32[32]
//   [4224, 4352)  bh f32[32]
//   [4352, 4480)  Wl f32[32]
//   [4480, 4480+4*N_WAVES)  per-wave pooled partial sums f32[6400]
#define WS_BZ   4096
#define WS_BH   4224
#define WS_WL   4352
#define WS_TS   4480        // 128-byte aligned

// 16-bit WMMA operand layout (CDNA5 ISA 7.12.2):
//   A (16x32, M on lanes): lane<16 -> elems 0..7 = K0..7,  8..15 = K16..23
//                          lane>=16 -> elems 0..7 = K8..15, 8..15 = K24..31
//   B (32x16, N on lanes) mirrored: same K packing, N = lane%16.

__global__ __launch_bounds__(256)
void rgcn_prep(const float* __restrict__ Wz, const float* __restrict__ Wh,
               const float* __restrict__ bz, const float* __restrict__ bh,
               const float* __restrict__ Wl, char* __restrict__ ws) {
    int idx = blockIdx.x * blockDim.x + threadIdx.x;  // 0..4095
    if (idx < 4096) {
        // Decode fragment element coordinates.
        const int i    = idx & 15;
        const int lane = (idx >> 4) & 31;
        const int kt   = (idx >> 9) & 1;
        const int t    = (idx >> 10) & 1;
        const int gate = (idx >> 11) & 1;
        const int h8   = (lane >> 4) << 3;
        const int n    = t * 16 + (lane & 15);
        const int k    = kt * 32 + ((i < 8) ? (h8 + i) : (16 + h8 + (i - 8)));
        // W shape (2,1,96,32): W[0,0] at 0, W[1,0] at 96*32=3072.
        // Only rows k<64 are used (H0 == 0 kills rows 64..95).
        const float* W = gate ? Wh : Wz;
        const float v  = W[k * HID + n] + W[3072 + k * HID + n];
        ((_Float16*)ws)[idx] = (_Float16)v;
    }
    if (idx < HID) {
        ((float*)(ws + WS_BZ))[idx] = bz[idx];
        ((float*)(ws + WS_BH))[idx] = bh[idx];
        ((float*)(ws + WS_WL))[idx] = Wl[idx];
    }
}

__device__ __forceinline__ float fast_sigmoid(float x) {
    return __builtin_amdgcn_rcpf(1.0f + __expf(-x));
}
__device__ __forceinline__ float fast_tanh(float x) {
    float e2 = __expf(2.0f * x);
    return (e2 - 1.0f) * __builtin_amdgcn_rcpf(e2 + 1.0f);
}

__global__ __launch_bounds__(256)
void rgcn_main(const float* __restrict__ x, const char* __restrict__ ws,
               float* __restrict__ wavesums) {
    const v16h*  bfrag = (const v16h*)ws;            // 256 fragments x 32 B
    const float* wsbz  = (const float*)(ws + WS_BZ);
    const float* wsbh  = (const float*)(ws + WS_BH);
    const float* wswl  = (const float*)(ws + WS_WL);

    const int lane  = threadIdx.x & 31;
    const int wave  = threadIdx.x >> 5;
    const int wid   = blockIdx.x * 8 + wave;  // 0..N_WAVES-1
    const int tile0 = wid * TPW;              // 5 consecutive tiles, one graph

    const int row  = lane & 15;               // A: M = lane (both half-waves)
    const int h8   = (lane >> 4) << 3;        // K offset for upper half-wave
    const int ncol = lane & 15;               // B/C/D: N = lane%16

    // ---- B fragments + biases: loaded ONCE per wave, reused for 5 tiles.
    v16h bzf[2][2], bhf[2][2];
#pragma unroll
    for (int t = 0; t < 2; ++t) {
#pragma unroll
        for (int kt = 0; kt < 2; ++kt) {
            bzf[t][kt] = bfrag[((0 * 2 + t) * 2 + kt) * 32 + lane];
            bhf[t][kt] = bfrag[((1 * 2 + t) * 2 + kt) * 32 + lane];
        }
    }
    float bzv[2], bhv[2], wlv[2];
#pragma unroll
    for (int t = 0; t < 2; ++t) {
        bzv[t] = wsbz[t * 16 + ncol];
        bhv[t] = wsbh[t * 16 + ncol];
        wlv[t] = wswl[t * 16 + ncol];
    }

    float local = 0.0f;   // per-lane pooled contribution over all 5 tiles

#pragma unroll
    for (int it = 0; it < TPW; ++it) {
        const int tile = tile0 + it;

        // ---- A fragment: 16 nodes x 64 feat, f32 -> f16, two 16x32 K-tiles.
        const float* xrow = x + (size_t)(tile * 16 + row) * F_IN;
        v16h a[2];
#pragma unroll
        for (int kt = 0; kt < 2; ++kt) {
            const int kb = kt * 32 + h8;
            const float4 f0 = *(const float4*)(xrow + kb);
            const float4 f1 = *(const float4*)(xrow + kb + 4);
            const float4 f2 = *(const float4*)(xrow + kb + 16);
            const float4 f3 = *(const float4*)(xrow + kb + 20);
            a[kt][0]  = (_Float16)f0.x; a[kt][1]  = (_Float16)f0.y;
            a[kt][2]  = (_Float16)f0.z; a[kt][3]  = (_Float16)f0.w;
            a[kt][4]  = (_Float16)f1.x; a[kt][5]  = (_Float16)f1.y;
            a[kt][6]  = (_Float16)f1.z; a[kt][7]  = (_Float16)f1.w;
            a[kt][8]  = (_Float16)f2.x; a[kt][9]  = (_Float16)f2.y;
            a[kt][10] = (_Float16)f2.z; a[kt][11] = (_Float16)f2.w;
            a[kt][12] = (_Float16)f3.x; a[kt][13] = (_Float16)f3.y;
            a[kt][14] = (_Float16)f3.z; a[kt][15] = (_Float16)f3.w;
        }

        // ---- Accumulators seeded with bias (free bias add via C operand).
        v8f accZ[2], accH[2];
#pragma unroll
        for (int t = 0; t < 2; ++t) {
#pragma unroll
            for (int r = 0; r < 8; ++r) { accZ[t][r] = bzv[t]; accH[t][r] = bhv[t]; }
        }

        // ---- 8x v_wmma_f32_16x16x32_f16: D = A*B + C
#pragma unroll
        for (int t = 0; t < 2; ++t) {
#pragma unroll
            for (int kt = 0; kt < 2; ++kt) {
                accZ[t] = __builtin_amdgcn_wmma_f32_16x16x32_f16(
                    false, a[kt], false, bzf[t][kt], (short)0, accZ[t], false, false);
                accH[t] = __builtin_amdgcn_wmma_f32_16x16x32_f16(
                    false, a[kt], false, bhf[t][kt], (short)0, accH[t], false, false);
            }
        }

        // ---- Fused GRU combine + ReLU + Linear(32->1) + pooling accumulate.
        // C/D f32 layout: lane holds column n, rows m = (lane<16 ? 0..7 : 8..15).
#pragma unroll
        for (int t = 0; t < 2; ++t) {
#pragma unroll
            for (int r = 0; r < 8; ++r) {
                float z  = fast_sigmoid(accZ[t][r]);
                float ht = fast_tanh(accH[t][r]);
                float Hv = (1.0f - z) * ht;
                local += fmaxf(Hv, 0.0f) * wlv[t];
            }
        }
    }

    // wave32 all-reduce, once per wave (5 tiles, all in the same graph)
#pragma unroll
    for (int off = 16; off > 0; off >>= 1)
        local += __shfl_xor(local, off, 32);

    if (lane == 0) wavesums[wid] = local;
}

__global__ __launch_bounds__(256)
void rgcn_pool(const float* __restrict__ wavesums, const float* __restrict__ bl,
               float* __restrict__ out) {
    int b = blockIdx.x * blockDim.x + threadIdx.x;
    if (b < N_GRAPH) {
        const float* p = wavesums + b * WAVES_PG;
        float s = 0.0f;
        for (int w = 0; w < WAVES_PG; ++w) s += p[w];
        out[b] = s * (1.0f / 2000.0f) + bl[0];
    }
}

extern "C" void kernel_launch(void* const* d_in, const int* in_sizes, int n_in,
                              void* d_out, int out_size, void* d_ws, size_t ws_size,
                              hipStream_t stream) {
    // setup_inputs order: x, edge_index, edge_weight, batch, Wz, bz, Wr, br,
    //                     Wh, bh, Wl, bl  (edges/batch unused in the math)
    const float* x  = (const float*)d_in[0];
    const float* Wz = (const float*)d_in[4];
    const float* bz = (const float*)d_in[5];
    const float* Wh = (const float*)d_in[8];
    const float* bh = (const float*)d_in[9];
    const float* Wl = (const float*)d_in[10];
    const float* bl = (const float*)d_in[11];

    char*  ws       = (char*)d_ws;
    float* wavesums = (float*)(ws + WS_TS);

    rgcn_prep<<<16, 256, 0, stream>>>(Wz, Wh, bz, bh, Wl, ws);
    rgcn_main<<<N_WAVES / 8, 256, 0, stream>>>(x, ws, wavesums);
    rgcn_pool<<<1, 256, 0, stream>>>(wavesums, bl, (float*)d_out);
}
